// MultiHeadAttention_31920196943920
// MI455X (gfx1250) — compile-verified, matrix operand order checked by reading, not run on hardware
//
#include <hip/hip_runtime.h>

// ---------------------------------------------------------------------------
// MultiHeadAttention forward for MI455X (gfx1250), wave32 + WMMA bf16 path.
// B=2, S=4096, E=768, H=12, D=64. Matmuls via v_wmma_f32_16x16x32_bf16.
// GEMM tiles are staged into LDS with the Tensor Data Mover (async, padded),
// double-buffered and synchronized with s_wait_tensorcnt.
// ---------------------------------------------------------------------------

#define SEQ    4096
#define EMB    768
#define HEADS  12
#define DEPTH  64
#define BATCH  2
#define ROWS   (BATCH * SEQ)      // 8192

typedef __attribute__((ext_vector_type(16))) __bf16    v16bf;
typedef __attribute__((ext_vector_type(8)))  float     v8f;
typedef __attribute__((ext_vector_type(4)))  int       v4i;
typedef __attribute__((ext_vector_type(8)))  int       v8i;
typedef __attribute__((ext_vector_type(4)))  unsigned  v4u;

union FragU {            // 32-byte WMMA operand fragment (16 x bf16)
    v4i   q[2];
    v16bf v;
};

__device__ __forceinline__ unsigned short f32_to_bf16(float f) {
    union { float f; unsigned u; } c; c.f = f;
    unsigned r = c.u + 0x7FFFu + ((c.u >> 16) & 1u);   // round-to-nearest-even
    return (unsigned short)(r >> 16);
}

__device__ __forceinline__ v8f vzero8() {
    return (v8f){0.f, 0.f, 0.f, 0.f, 0.f, 0.f, 0.f, 0.f};
}

// ---------------------------------------------------------------------------
// TDM: async 2-D bf16 tile load global -> LDS with row padding.
// pad_interval = 16 dwords (one 32-elem bf16 row), pad_amount = 4 dwords
// => LDS row stride 20 dwords = 80 B = 40 bf16 elems (bank-conflict-free).
// ---------------------------------------------------------------------------
#define LSTR 40   // padded LDS row stride in bf16 elems

__device__ __forceinline__ void tdm_load_2d_bf16(
    const unsigned short* gptr,  // global address of tile start
    unsigned lds_off,            // LDS byte offset of tile start
    unsigned tensor_d0,          // tensor dim0 (elems)  = K
    unsigned tensor_d1,          // tensor dim1 (rows)
    unsigned tile_d0,            // tile dim0  (elems)   = 32
    unsigned tile_d1,            // tile dim1  (rows)
    unsigned stride0)            // row stride (elems)   = K
{
    unsigned long long ga = (unsigned long long)(uintptr_t)gptr;
    // D# group 0: count=1 valid, lds_addr, global_addr[56:0], type=2
    v4u g0 = { 1u,
               lds_off,
               (unsigned)ga,
               ((unsigned)(ga >> 32) & 0x01FFFFFFu) | (2u << 30) };
    // D# group 1: data_size=2B, pad_enable, pad_interval=16dw, pad_amount=4dw
    const unsigned w0 = 0x10000u | (1u << 20) | (3u << 22) | (3u << 25);
    v8i g1 = { (int)w0,
               (int)((tensor_d0 & 0xFFFFu) << 16),                    // dim0 lo
               (int)((tensor_d0 >> 16) | ((tensor_d1 & 0xFFFFu) << 16)),
               (int)((tensor_d1 >> 16) | (tile_d0 << 16)),
               (int)(tile_d1 & 0xFFFFu),                              // tile_d2=0
               (int)stride0,                                          // stride lo32
               0, 0 };                                                // stride hi / dim1_stride
    v4i z4 = {0, 0, 0, 0};
#if __clang_major__ >= 23
    v8i z8 = {0, 0, 0, 0, 0, 0, 0, 0};
    __builtin_amdgcn_tensor_load_to_lds(g0, g1, z4, z4, z8, 0);
#else
    __builtin_amdgcn_tensor_load_to_lds(g0, g1, z4, z4, 0);
#endif
}

// ---------------------------------------------------------------------------
// fp32 -> bf16 elementwise conversion
// ---------------------------------------------------------------------------
__global__ void __launch_bounds__(256)
cvt_bf16_kernel(const float* __restrict__ in, unsigned short* __restrict__ out, int n) {
    int i = blockIdx.x * 256 + threadIdx.x;
    if (i < n) out[i] = f32_to_bf16(in[i]);
}

// ---------------------------------------------------------------------------
// Generic GEMM:  C[m,n] = sum_k A[m,k] * W[n,k]    (torch Linear semantics)
// A: bf16 [M x K] row-major, W: bf16 [N x K] row-major.
// mode 0: bf16 out, per-head row-major  [(b*H+h)][s][d]      (Q, K)
// mode 1: bf16 out, per-head transposed [(b*H+h)][d][s]      (V)
// mode 2: fp32 out [M x N] + bias[n]                         (final proj)
// Tiling: BM=256 (8 waves x 32 rows), BN=64 (4 n-tiles), BK=32.
// TDM double-buffered staging: wave w DMAs A rows [w*32,w*32+32) and
// W rows [w*8,w*8+8) of the next K-tile while computing the current one.
// ---------------------------------------------------------------------------
#define BM 256
#define BN 64
#define BK 32

__global__ void __launch_bounds__(256)
gemm_bf16_wmma(const unsigned short* __restrict__ A,
               const unsigned short* __restrict__ W,
               const float* __restrict__ bias,
               unsigned short* __restrict__ outB,
               float* __restrict__ outF,
               int M, int N, int K, int mode)
{
    __shared__ __align__(16) unsigned short As[2 * BM * LSTR];   // 40 KB
    __shared__ __align__(16) unsigned short Bs[2 * BN * LSTR];   // 10 KB

    const int tid  = threadIdx.x;
    const int wave = tid >> 5;
    const int lane = tid & 31;
    const int lrow = lane & 15;
    const int half = lane >> 4;
    const int m0   = blockIdx.x * BM;
    const int n0   = blockIdx.y * BN;
    const int wv   = __builtin_amdgcn_readfirstlane(wave);   // scalar wave id

    const unsigned lds_a = (unsigned)(uintptr_t)(void*)As;   // LDS byte offsets
    const unsigned lds_b = (unsigned)(uintptr_t)(void*)Bs;

    v8f acc[2][4];
#pragma unroll
    for (int mt = 0; mt < 2; ++mt)
#pragma unroll
        for (int nt = 0; nt < 4; ++nt) acc[mt][nt] = vzero8();

    // prologue: DMA first K-tile into buffer 0
    tdm_load_2d_bf16(A + (size_t)(m0 + wv * 32) * K, lds_a + (unsigned)(wv * 32 * LSTR * 2),
                     (unsigned)K, (unsigned)M, BK, 32, (unsigned)K);
    tdm_load_2d_bf16(W + (size_t)(n0 + wv * 8) * K,  lds_b + (unsigned)(wv * 8 * LSTR * 2),
                     (unsigned)K, (unsigned)N, BK, 8,  (unsigned)K);

    int it = 0;
    for (int k0 = 0; k0 < K; k0 += BK, ++it) {
        const int cur = it & 1;
        const int nxt = cur ^ 1;
        const bool more = (k0 + BK) < K;
        if (more) {   // DMA next K-tile into the other buffer
            tdm_load_2d_bf16(A + (size_t)(m0 + wv * 32) * K + (k0 + BK),
                             lds_a + (unsigned)((nxt * BM + wv * 32) * LSTR * 2),
                             (unsigned)K, (unsigned)M, BK, 32, (unsigned)K);
            tdm_load_2d_bf16(W + (size_t)(n0 + wv * 8) * K + (k0 + BK),
                             lds_b + (unsigned)((nxt * BN + wv * 8) * LSTR * 2),
                             (unsigned)K, (unsigned)N, BK, 8,  (unsigned)K);
            __builtin_amdgcn_s_wait_tensorcnt(2);   // current tile complete
        } else {
            __builtin_amdgcn_s_wait_tensorcnt(0);
        }
        __syncthreads();   // publish current buffer to all waves

        // preload all fragments, then issue the WMMA burst back-to-back
        FragU a[2], b[4];
#pragma unroll
        for (int mt = 0; mt < 2; ++mt) {
            const unsigned short* ap =
                As + (cur * BM + wave * 32 + mt * 16 + lrow) * LSTR + half * 8;
            a[mt].q[0] = *(const v4i*)ap;
            a[mt].q[1] = *(const v4i*)(ap + 16);
        }
#pragma unroll
        for (int nt = 0; nt < 4; ++nt) {
            const unsigned short* bp =
                Bs + (cur * BN + nt * 16 + lrow) * LSTR + half * 8;
            b[nt].q[0] = *(const v4i*)bp;
            b[nt].q[1] = *(const v4i*)(bp + 16);
        }
#pragma unroll
        for (int mt = 0; mt < 2; ++mt)
#pragma unroll
            for (int nt = 0; nt < 4; ++nt)
                acc[mt][nt] = __builtin_amdgcn_wmma_f32_16x16x32_bf16(
                    false, a[mt].v, false, b[nt].v, (short)0, acc[mt][nt], false, false);

        __syncthreads();   // done reading current buffer before it is re-DMAed
    }

    // --- epilogue: C layout = VGPR r, lanes 0-15: M=r, lanes 16-31: M=8+r ---
#pragma unroll
    for (int mt = 0; mt < 2; ++mt) {
        const int mrow = m0 + wave * 32 + mt * 16;
#pragma unroll
        for (int nt = 0; nt < 4; ++nt) {
#pragma unroll
            for (int r = 0; r < 8; ++r) {
                int   m = mrow + r + 8 * half;
                int   n = n0 + nt * 16 + lrow;
                float v = acc[mt][nt][r];
                if (mode == 2) {
                    outF[(size_t)m * N + n] = v + bias[n];
                } else {
                    int b2 = m >> 12, s = m & (SEQ - 1);
                    int h  = n >> 6,  d = n & (DEPTH - 1);
                    size_t idx = (mode == 0)
                        ? (((size_t)(b2 * HEADS + h) * SEQ + s) * DEPTH + d)
                        : (((size_t)(b2 * HEADS + h) * DEPTH + d) * SEQ + s);
                    outB[idx] = f32_to_bf16(v);
                }
            }
        }
    }
}

// ---------------------------------------------------------------------------
// Flash attention forward. Grid: (B*H, S/128). Block: 256 = 8 waves.
// Each wave owns a 16-row Q tile; iterates key blocks of 64.
// Q,K: bf16 [BH][S][64]   V: bf16 [BH][64][S] (transposed)
// O: bf16 [B*S][768] (heads re-interleaved for the output projection)
// ---------------------------------------------------------------------------
__global__ void __launch_bounds__(256)
attn_fwd_wmma(const unsigned short* __restrict__ Q,
              const unsigned short* __restrict__ Km,
              const unsigned short* __restrict__ Vt,
              unsigned short* __restrict__ O)
{
    __shared__ __align__(16) unsigned short Plds[8 * 16 * 64];  // 16 KB

    const int tid  = threadIdx.x;
    const int wave = tid >> 5;
    const int lane = tid & 31;
    const int lrow = lane & 15;
    const int half = lane >> 4;
    const int bh   = blockIdx.x;                 // 0..23
    const int m0   = blockIdx.y * 128 + wave * 16;

    const unsigned short* Qh = Q  + (size_t)bh * SEQ * DEPTH;
    const unsigned short* Kh = Km + (size_t)bh * SEQ * DEPTH;
    const unsigned short* Vh = Vt + (size_t)bh * DEPTH * SEQ;

    // Q fragments (D=64 -> two 16x32 A-operands), loaded once
    FragU qf[2];
#pragma unroll
    for (int kt = 0; kt < 2; ++kt) {
        const unsigned short* p = Qh + (size_t)(m0 + lrow) * DEPTH + kt * 32 + half * 8;
        qf[kt].q[0] = *(const v4i*)p;
        qf[kt].q[1] = *(const v4i*)(p + 16);
    }

    float mrun[8], lrun[8];
    v8f   oacc[4];
#pragma unroll
    for (int r = 0; r < 8; ++r) { mrun[r] = -1e30f; lrun[r] = 0.f; }
#pragma unroll
    for (int dt = 0; dt < 4; ++dt) oacc[dt] = vzero8();

    unsigned short* Pw = Plds + wave * (16 * 64);

    for (int j0 = 0; j0 < SEQ; j0 += 64) {
        // ---- S = (Q K^T) * 1/sqrt(D) : 4 n-tiles x 2 K-steps = 8 WMMAs ----
        v8f sacc[4];
#pragma unroll
        for (int nt = 0; nt < 4; ++nt) {
            const unsigned short* kp = Kh + (size_t)(j0 + nt * 16 + lrow) * DEPTH + half * 8;
            FragU k0f, k1f;
            k0f.q[0] = *(const v4i*)kp;         k0f.q[1] = *(const v4i*)(kp + 16);
            k1f.q[0] = *(const v4i*)(kp + 32);  k1f.q[1] = *(const v4i*)(kp + 48);
            v8f c = vzero8();
            c = __builtin_amdgcn_wmma_f32_16x16x32_bf16(false, qf[0].v, false, k0f.v, (short)0, c, false, false);
            c = __builtin_amdgcn_wmma_f32_16x16x32_bf16(false, qf[1].v, false, k1f.v, (short)0, c, false, false);
            sacc[nt] = c * 0.125f;   // 1/sqrt(64)
        }

        // ---- online softmax: rows live within a 16-lane half ----
        float alpha[8];
#pragma unroll
        for (int r = 0; r < 8; ++r) {
            float mx = fmaxf(fmaxf(sacc[0][r], sacc[1][r]), fmaxf(sacc[2][r], sacc[3][r]));
#pragma unroll
            for (int off = 8; off >= 1; off >>= 1)
                mx = fmaxf(mx, __shfl_xor(mx, off, 32));
            float nm = fmaxf(mrun[r], mx);
            alpha[r] = __expf(mrun[r] - nm);
            mrun[r]  = nm;
        }
        float rsum[8];
#pragma unroll
        for (int r = 0; r < 8; ++r) rsum[r] = 0.f;
#pragma unroll
        for (int nt = 0; nt < 4; ++nt) {
#pragma unroll
            for (int r = 0; r < 8; ++r) {
                float p = __expf(sacc[nt][r] - mrun[r]);
                sacc[nt][r] = p;
                rsum[r] += p;
            }
        }
#pragma unroll
        for (int r = 0; r < 8; ++r) {
            float s = rsum[r];
#pragma unroll
            for (int off = 8; off >= 1; off >>= 1)
                s += __shfl_xor(s, off, 32);
            lrun[r] = lrun[r] * alpha[r] + s;
        }
#pragma unroll
        for (int dt = 0; dt < 4; ++dt)
#pragma unroll
            for (int r = 0; r < 8; ++r) oacc[dt][r] *= alpha[r];

        // ---- C-layout -> A-layout transpose of P via per-wave LDS ----
#pragma unroll
        for (int nt = 0; nt < 4; ++nt)
#pragma unroll
            for (int r = 0; r < 8; ++r)
                Pw[(r + 8 * half) * 64 + nt * 16 + lrow] = f32_to_bf16(sacc[nt][r]);
        asm volatile("s_wait_dscnt 0" ::: "memory");   // wave-local LDS RAW fence

        FragU pf[2];
#pragma unroll
        for (int kk = 0; kk < 2; ++kk) {
            const unsigned short* pp = Pw + lrow * 64 + kk * 32 + half * 8;
            pf[kk].q[0] = *(const v4i*)pp;
            pf[kk].q[1] = *(const v4i*)(pp + 16);
        }

        // ---- O += P V : 4 d-tiles x 2 key-K-steps = 8 WMMAs ----
#pragma unroll
        for (int dt = 0; dt < 4; ++dt) {
#pragma unroll
            for (int kk = 0; kk < 2; ++kk) {
                const unsigned short* vp =
                    Vh + (size_t)(dt * 16 + lrow) * SEQ + j0 + kk * 32 + half * 8;
                FragU vf;
                vf.q[0] = *(const v4i*)vp;
                vf.q[1] = *(const v4i*)(vp + 16);
                oacc[dt] = __builtin_amdgcn_wmma_f32_16x16x32_bf16(
                    false, pf[kk].v, false, vf.v, (short)0, oacc[dt], false, false);
            }
        }
    }

    // ---- epilogue: normalize by l, scatter heads back to [B*S][768] ----
    const int b = bh / HEADS, h = bh % HEADS;
    float inv[8];
#pragma unroll
    for (int r = 0; r < 8; ++r) inv[r] = 1.0f / lrun[r];
#pragma unroll
    for (int dt = 0; dt < 4; ++dt) {
#pragma unroll
        for (int r = 0; r < 8; ++r) {
            int s = m0 + r + 8 * half;
            int d = dt * 16 + lrow;
            O[((size_t)(b * SEQ + s)) * EMB + h * DEPTH + d] =
                f32_to_bf16(oacc[dt][r] * inv[r]);
        }
    }
}

// ---------------------------------------------------------------------------
// Host launch
// ---------------------------------------------------------------------------
extern "C" void kernel_launch(void* const* d_in, const int* in_sizes, int n_in,
                              void* d_out, int out_size, void* d_ws, size_t ws_size,
                              hipStream_t stream) {
    const float* x   = (const float*)d_in[0];
    const float* Wq  = (const float*)d_in[1];
    const float* Wk  = (const float*)d_in[2];
    const float* Wv  = (const float*)d_in[3];
    const float* Wo  = (const float*)d_in[4];
    const float* bo  = (const float*)d_in[5];
    float*       out = (float*)d_out;

    char*  ws  = (char*)d_ws;
    size_t off = 0;
    auto alloc_u16 = [&](size_t elems) -> unsigned short* {
        unsigned short* p = (unsigned short*)(ws + off);
        off = (off + elems * sizeof(unsigned short) + 255) & ~(size_t)255;
        return p;
    };

    const size_t nX = (size_t)ROWS * EMB;                   // 6,291,456
    const size_t nW = (size_t)EMB * EMB;                    //   589,824
    const size_t nH = (size_t)BATCH * HEADS * SEQ * DEPTH;  // 6,291,456

    unsigned short* Xb  = alloc_u16(nX);
    unsigned short* Wqb = alloc_u16(nW);
    unsigned short* Wkb = alloc_u16(nW);
    unsigned short* Wvb = alloc_u16(nW);
    unsigned short* Wob = alloc_u16(nW);
    unsigned short* Qb  = alloc_u16(nH);
    unsigned short* Kb  = alloc_u16(nH);
    unsigned short* Vtb = alloc_u16(nH);
    unsigned short* Ob  = alloc_u16(nX);
    (void)ws_size; (void)in_sizes; (void)n_in; (void)out_size;

    // 1) fp32 -> bf16 conversions
    cvt_bf16_kernel<<<(int)((nX + 255) / 256), 256, 0, stream>>>(x,  Xb,  (int)nX);
    cvt_bf16_kernel<<<(int)((nW + 255) / 256), 256, 0, stream>>>(Wq, Wqb, (int)nW);
    cvt_bf16_kernel<<<(int)((nW + 255) / 256), 256, 0, stream>>>(Wk, Wkb, (int)nW);
    cvt_bf16_kernel<<<(int)((nW + 255) / 256), 256, 0, stream>>>(Wv, Wvb, (int)nW);
    cvt_bf16_kernel<<<(int)((nW + 255) / 256), 256, 0, stream>>>(Wo, Wob, (int)nW);

    // 2) QKV projections (x @ W^T), scattered to per-head layouts
    dim3 gg(ROWS / BM, EMB / BN);   // (32, 12)
    gemm_bf16_wmma<<<gg, 256, 0, stream>>>(Xb, Wqb, nullptr, Qb,  nullptr, ROWS, EMB, EMB, 0);
    gemm_bf16_wmma<<<gg, 256, 0, stream>>>(Xb, Wkb, nullptr, Kb,  nullptr, ROWS, EMB, EMB, 0);
    gemm_bf16_wmma<<<gg, 256, 0, stream>>>(Xb, Wvb, nullptr, Vtb, nullptr, ROWS, EMB, EMB, 1);

    // 3) flash attention
    dim3 ga(BATCH * HEADS, SEQ / 128);   // (24, 32)
    attn_fwd_wmma<<<ga, 256, 0, stream>>>(Qb, Kb, Vtb, Ob);

    // 4) output projection + bias, fp32 out
    gemm_bf16_wmma<<<gg, 256, 0, stream>>>(Ob, Wob, bo, nullptr, out, ROWS, EMB, EMB, 2);
}
